// FaceRecon_36721970380938
// MI455X (gfx1250) — matrix operand order repaired and checked by reading, not compiled, not run on hardware
//
#include <hip/hip_runtime.h>
#include <hip/hip_bf16.h>
#include <math.h>

typedef __attribute__((ext_vector_type(16))) _Float16 v16h;
typedef __attribute__((ext_vector_type(8)))  float    v8f;

#define CDIV(a,b) (((a)+(b)-1)/(b))

// ======================= weight pre-pack =======================
// W16[Npad x Kpad] f16, zero padded.  wt=1: src W[N,K] ("oc"); wt=0: src W[K,N] ("cm").
__global__ void wprep_kernel(const float* __restrict__ W, _Float16* __restrict__ W16,
                             int N, int K, int Kpad, int wt, int total)
{
  int i = blockIdx.x * blockDim.x + threadIdx.x;
  if (i >= total) return;
  int n = i / Kpad, k = i - n * Kpad;
  float v = 0.f;
  if (n < N && k < K) v = wt ? W[(size_t)n*K + k] : W[(size_t)k*N + n];
  W16[i] = (_Float16)v;
}

// ======================= WMMA GEMM =======================
// C[M,N] = act( A[M,K] @ B + bias ); A f32, B pre-packed f16 [Npad x Kpad].
// Wave computes a 16x64 strip (4 WMMA tiles, A fragment reused 4x);
// 4 waves/block cover 64 rows. Main loop is branch-free wide loads + 4 WMMAs.
// VEC=1: K%32==0 and A rows 32B-aligned -> v8f (b128) A loads.
template<int VEC>
__global__ __launch_bounds__(128)
void gemm_wmma_kernel(const float* __restrict__ A, const _Float16* __restrict__ W16,
                      const float* __restrict__ bias, float* __restrict__ C,
                      int M, int N, int K, int Kpad, int relu)
{
  const int lane = threadIdx.x & 31;
  const int wave = threadIdx.x >> 5;
  const int half = lane >> 4;      // K-half of the wave
  const int l15  = lane & 15;
  const int m0 = blockIdx.y * 64 + wave * 16;
  const int n0 = blockIdx.x * 64;
  const int mrow = m0 + l15;       // both halves hold rows 0..15 of the tile
  v8f acc0 = {}, acc1 = {}, acc2 = {}, acc3 = {};
  const int Kfloor = K & ~31;
  const _Float16* wp = W16 + (size_t)(n0 + l15) * Kpad + half * 16;
  const size_t wstep = (size_t)16 * Kpad;
  const float* ap = A + (size_t)mrow * K + half * 16;

  for (int kk = 0; kk < Kfloor; kk += 32) {
    v16h a;
    if (VEC) {
      v8f g0 = *(const v8f*)(ap + kk);         // K = half*16 .. +7
      v8f g1 = *(const v8f*)(ap + kk + 8);     // K = half*16+8 .. +7
#pragma unroll
      for (int e = 0; e < 8; ++e) { a[e] = (_Float16)g0[e]; a[8+e] = (_Float16)g1[e]; }
    } else {
#pragma unroll
      for (int e = 0; e < 8; ++e) { a[e] = (_Float16)ap[kk+e]; a[8+e] = (_Float16)ap[kk+8+e]; }
    }
    v16h b0 = *(const v16h*)(wp + kk);             // col n0+l15,  K = kk+half*16..+15
    v16h b1 = *(const v16h*)(wp + wstep   + kk);   // col n0+16+l15
    v16h b2 = *(const v16h*)(wp + wstep*2 + kk);   // col n0+32+l15
    v16h b3 = *(const v16h*)(wp + wstep*3 + kk);   // col n0+48+l15
    acc0 = __builtin_amdgcn_wmma_f32_16x16x32_f16(false, a, false, b0, (short)0, acc0, false, false);
    acc1 = __builtin_amdgcn_wmma_f32_16x16x32_f16(false, a, false, b1, (short)0, acc1, false, false);
    acc2 = __builtin_amdgcn_wmma_f32_16x16x32_f16(false, a, false, b2, (short)0, acc2, false, false);
    acc3 = __builtin_amdgcn_wmma_f32_16x16x32_f16(false, a, false, b3, (short)0, acc3, false, false);
  }
  if (Kfloor < K) {                // ragged K tail (K=1286/771 only), one guarded step
    v16h a;
#pragma unroll
    for (int e = 0; e < 16; ++e) {
      int ka = half * 16 + e;
      int k = Kfloor + ka;
      float av = (k < K) ? A[(size_t)mrow * K + k] : 0.f;
      a[e] = (_Float16)av;
    }
    v16h b0 = *(const v16h*)(wp + Kfloor);         // W16 zero-padded to Kpad
    v16h b1 = *(const v16h*)(wp + wstep   + Kfloor);
    v16h b2 = *(const v16h*)(wp + wstep*2 + Kfloor);
    v16h b3 = *(const v16h*)(wp + wstep*3 + Kfloor);
    acc0 = __builtin_amdgcn_wmma_f32_16x16x32_f16(false, a, false, b0, (short)0, acc0, false, false);
    acc1 = __builtin_amdgcn_wmma_f32_16x16x32_f16(false, a, false, b1, (short)0, acc1, false, false);
    acc2 = __builtin_amdgcn_wmma_f32_16x16x32_f16(false, a, false, b2, (short)0, acc2, false, false);
    acc3 = __builtin_amdgcn_wmma_f32_16x16x32_f16(false, a, false, b3, (short)0, acc3, false, false);
  }
#pragma unroll
  for (int r = 0; r < 8; ++r) {    // C layout: lane col = l15, row = half*8 + r
    int m = m0 + half * 8 + r;
    if (m >= M) continue;
    float* crow = C + (size_t)m * N;
#pragma unroll
    for (int j = 0; j < 4; ++j) {
      int n = n0 + j * 16 + l15;
      if (n < N) {
        float v = (j == 0) ? acc0[r] : (j == 1) ? acc1[r] : (j == 2) ? acc2[r] : acc3[r];
        if (bias) v += bias[n];
        if (relu) v = fmaxf(v, 0.f);
        crow[n] = v;
      }
    }
  }
}

// ======================= kNN top-(K+1), stable ties, drop self =======================
template<int KP1>
__global__ void knn_kernel(const float* __restrict__ verts, int* __restrict__ idx,
                           int B, int Ns, int Nq, int qstride)
{
  int t = blockIdx.x * blockDim.x + threadIdx.x;
  if (t >= B * Nq) return;
  int b = t / Nq, nq = t - b * Nq;
  const float* base = verts + (size_t)b * Ns * 3;
  int qi = nq * qstride;
  float qx = base[qi*3+0], qy = base[qi*3+1], qz = base[qi*3+2];
  float bd[KP1]; int bi[KP1];
#pragma unroll
  for (int j = 0; j < KP1; ++j) { bd[j] = 3.4e38f; bi[j] = 0; }
  for (int sp = 0; sp < Ns; ++sp) {
    float dx = base[sp*3+0]-qx, dy = base[sp*3+1]-qy, dz = base[sp*3+2]-qz;
    float d = dx*dx + dy*dy + dz*dz;
    if (d < bd[KP1-1]) {
#pragma unroll
      for (int j = KP1-1; j >= 0; --j) {     // register insertion (no dynamic indexing)
        bool shift = (j > 0) && (d < bd[j-1]);
        if (shift)            { bd[j] = bd[j-1]; bi[j] = bi[j-1]; }
        else if (d < bd[j])   { bd[j] = d;       bi[j] = sp;      }
      }
    }
  }
#pragma unroll
  for (int j = 1; j < KP1; ++j)              // drop nearest (self)
    idx[(size_t)t * (KP1-1) + (j-1)] = bi[j];
}

// ======================= normalize dirs over axis 0 =======================
__global__ void ndirs_kernel(const float* __restrict__ dirs, float* __restrict__ out, int Mc)
{
  int m = blockIdx.x * blockDim.x + threadIdx.x;
  if (m >= Mc) return;
  float a = dirs[m], b = dirs[Mc + m], c = dirs[2*Mc + m];
  float n = fmaxf(sqrtf(a*a + b*b + c*c), 1e-12f);
  out[m] = a/n; out[Mc + m] = b/n; out[2*Mc + m] = c/n;
}

// ======================= surface / layer local feature =======================
__global__ __launch_bounds__(128)
void hs_local_kernel(const float* __restrict__ verts, const int* __restrict__ idx,
                     const float* __restrict__ ndirs, const float* __restrict__ ste,
                     const float* __restrict__ fo, float* __restrict__ out,
                     int Npts, int K, int cout)
{
  __shared__ float snd[10*3];
  __shared__ int   snbr[10];
  int row = blockIdx.x;
  int b = row / Npts;
  float vx = verts[(size_t)row*3+0], vy = verts[(size_t)row*3+1], vz = verts[(size_t)row*3+2];
  if ((int)threadIdx.x < K) {
    int nb = idx[(size_t)row*K + threadIdx.x];
    int nrow = b * Npts + nb;
    float dx = verts[(size_t)nrow*3+0]-vx, dy = verts[(size_t)nrow*3+1]-vy, dz = verts[(size_t)nrow*3+2]-vz;
    float n = fmaxf(sqrtf(dx*dx+dy*dy+dz*dz), 1e-12f);
    snd[threadIdx.x*3+0]=dx/n; snd[threadIdx.x*3+1]=dy/n; snd[threadIdx.x*3+2]=dz/n;
    snbr[threadIdx.x]=nrow;
  }
  __syncthreads();
  const int SC = 7 * cout;
  const bool hasfo = (fo != nullptr);
  const int fstride = 8 * cout;
  for (int o = threadIdx.x; o < cout; o += blockDim.x) {
    float acc = vx*ste[o*3+0] + vy*ste[o*3+1] + vz*ste[o*3+2];   // f_ste
    if (hasfo) acc += fo[(size_t)row*fstride + o];               // f_center
    for (int s7 = 0; s7 < 7; ++s7) {
      int m = s7*cout + o;
      float c0 = ndirs[m], c1 = ndirs[SC + m], c2 = ndirs[2*SC + m];
      float mx = hasfo ? -3.4e38f : 0.f;
      for (int k = 0; k < K; ++k) {
        float t = fmaxf(snd[k*3+0]*c0 + snd[k*3+1]*c1 + snd[k*3+2]*c2, 0.f);
        if (hasfo) t *= fo[(size_t)snbr[k]*fstride + cout + m];  // gathered f_support
        mx = fmaxf(mx, t);
      }
      acc += mx;
    }
    out[(size_t)row*cout + o] = acc;
  }
}

// ======================= elementwise / gather helpers =======================
__global__ void gathermean_kernel(const float* __restrict__ f, const int* __restrict__ idx,
                                  float* __restrict__ out, int Npts, int K, int C, int total)
{
  int i = blockIdx.x * blockDim.x + threadIdx.x;
  if (i >= total) return;
  int row = i / C, c = i - row * C;
  int b = row / Npts;
  const int* ir = idx + (size_t)row * K;
  float s = 0.f;
  for (int k = 0; k < K; ++k) s += f[((size_t)b*Npts + ir[k])*C + c];
  out[i] = s / (float)K;
}

__global__ void concat2_kernel(const float* __restrict__ x1, const float* __restrict__ x2,
                               float* __restrict__ out, int C1, int C2, int total)
{
  int i = blockIdx.x * blockDim.x + threadIdx.x;
  if (i >= total) return;
  int C = C1 + C2; int row = i / C, c = i - row * C;
  out[i] = (c < C1) ? x1[(size_t)row*C1 + c] : x2[(size_t)row*C2 + (c - C1)];
}

__global__ void meanvar_kernel(const float* __restrict__ x, float* __restrict__ mean,
                               float* __restrict__ var, int M, int C)
{
  __shared__ float ss[256], ss2[256];
  int c = blockIdx.x;
  float s = 0.f, s2 = 0.f;
  for (int m = threadIdx.x; m < M; m += blockDim.x) {
    float v = x[(size_t)m*C + c]; s += v; s2 += v*v;
  }
  ss[threadIdx.x] = s; ss2[threadIdx.x] = s2;
  __syncthreads();
  for (int st = 128; st > 0; st >>= 1) {
    if ((int)threadIdx.x < st) { ss[threadIdx.x] += ss[threadIdx.x+st]; ss2[threadIdx.x] += ss2[threadIdx.x+st]; }
    __syncthreads();
  }
  if (threadIdx.x == 0) {
    float mu = ss[0] / (float)M;
    mean[c] = mu;
    var[c] = ss2[0] / (float)M - mu * mu;
  }
}

__global__ void bn_kernel(const float* __restrict__ x, float* __restrict__ y,
                          const float* __restrict__ mean, const float* __restrict__ var,
                          const float* __restrict__ g, const float* __restrict__ b,
                          int C, int relu, int total)
{
  int i = blockIdx.x * blockDim.x + threadIdx.x;
  if (i >= total) return;
  int c = i % C;
  float v = g[c] * (x[i] - mean[c]) * rsqrtf(var[c] + 1e-5f) + b[c];
  y[i] = relu ? fmaxf(v, 0.f) : v;
}

__global__ void poolmax_kernel(const float* __restrict__ f, const int* __restrict__ idx,
                               float* __restrict__ out, int Ns, int K, int C, int Nq, int total)
{
  int i = blockIdx.x * blockDim.x + threadIdx.x;
  if (i >= total) return;
  int row = i / C, c = i - row * C;
  int b = row / Nq;
  float m = -3.4e38f;
  for (int k = 0; k < K; ++k)
    m = fmaxf(m, f[((size_t)b*Ns + idx[(size_t)row*K + k])*C + c]);
  out[i] = m;
}

__global__ void subsample_kernel(const float* __restrict__ v, float* __restrict__ o,
                                 int Ns, int Nq, int rate, int total)
{
  int i = blockIdx.x * blockDim.x + threadIdx.x;
  if (i >= total) return;
  int row = i / 3, d = i - row * 3;
  int b = row / Nq, j = row - b * Nq;
  o[i] = v[((size_t)b*Ns + (size_t)j*rate)*3 + d];
}

__global__ void nearest_kernel(const float* __restrict__ tgt, const float* __restrict__ src,
                               int* __restrict__ out, int Nt, int Ns, int total)
{
  int t = blockIdx.x * blockDim.x + threadIdx.x;
  if (t >= total) return;
  int b = t / Nt;
  float qx = tgt[(size_t)t*3+0], qy = tgt[(size_t)t*3+1], qz = tgt[(size_t)t*3+2];
  const float* base = src + (size_t)b * Ns * 3;
  float bestd = 3.4e38f; int besti = 0;
  for (int sp = 0; sp < Ns; ++sp) {
    float dx = base[sp*3+0]-qx, dy = base[sp*3+1]-qy, dz = base[sp*3+2]-qz;
    float d = dx*dx + dy*dy + dz*dz;
    if (d < bestd) { bestd = d; besti = sp; }
  }
  out[t] = besti;
}

__global__ void gmax_kernel(const float* __restrict__ f, float* __restrict__ out,
                            int Npts, int C, int total)
{
  int i = blockIdx.x * blockDim.x + threadIdx.x;
  if (i >= total) return;
  int b = i / C, c = i - b * C;
  float m = -3.4e38f;
  for (int n = 0; n < Npts; ++n)
    m = fmaxf(m, f[((size_t)b*Npts + n)*C + c]);
  out[i] = m;
}

__global__ void feat_kernel(const float* __restrict__ fm0, const float* __restrict__ fm1,
                            const float* __restrict__ fm2, const float* __restrict__ fm3,
                            const float* __restrict__ fm4,
                            const int* __restrict__ n1, const int* __restrict__ n2,
                            const int* __restrict__ cat, float* __restrict__ out,
                            int N0, int N1p, int N2p, int total)
{
  int i = blockIdx.x * blockDim.x + threadIdx.x;
  if (i >= total) return;
  int row = i / 1286, c = i - row * 1286;
  int b = row / N0;
  float v;
  if (c < 128)        v = fm0[(size_t)row*128 + c];
  else if (c < 256)   v = fm1[(size_t)row*128 + (c-128)];
  else if (c < 512)   v = fm2[((size_t)b*N1p + n1[row])*256 + (c-256)];
  else if (c < 768)   v = fm3[((size_t)b*N1p + n1[row])*256 + (c-512)];
  else if (c < 1280)  v = fm4[((size_t)b*N2p + n2[row])*512 + (c-768)];
  else                v = (cat[b] == (c - 1280)) ? 1.f : 0.f;
  out[i] = v;
}

__global__ void facein_kernel(const float* __restrict__ fglob, const float* __restrict__ x,
                              const float* __restrict__ verts, float* __restrict__ out,
                              int N0, int total)
{
  int i = blockIdx.x * blockDim.x + threadIdx.x;
  if (i >= total) return;
  int row = i / 771, c = i - row * 771;
  int b = row / N0;
  float v;
  if (c < 512)      v = fglob[(size_t)b*512 + c];
  else if (c < 768) v = x[(size_t)row*256 + (c-512)];
  else              v = verts[(size_t)row*3 + (c-768)];
  out[i] = v;
}

// ======================= host-side helpers =======================
static void gemm(const float* A, const float* Wm, const float* bias, float* C,
                 int M, int N, int K, int wt, int relu, _Float16* wbuf, hipStream_t s)
{
  int Kpad = (K + 31) & ~31;
  int Npad = (N + 63) & ~63;
  wprep_kernel<<<CDIV(Npad*Kpad,256), 256, 0, s>>>(Wm, wbuf, N, K, Kpad, wt, Npad*Kpad);
  dim3 g(Npad/64, CDIV(M,64));
  if ((K & 31) == 0)
    gemm_wmma_kernel<1><<<g, 128, 0, s>>>(A, wbuf, bias, C, M, N, K, Kpad, relu);
  else
    gemm_wmma_kernel<0><<<g, 128, 0, s>>>(A, wbuf, bias, C, M, N, K, Kpad, relu);
}

static void run_bn(float* x, float* y, const float* g, const float* b,
                   float* mean, float* var, int M, int C, int relu, hipStream_t s)
{
  meanvar_kernel<<<C, 256, 0, s>>>(x, mean, var, M, C);
  bn_kernel<<<CDIV(M*C,256), 256, 0, s>>>(x, y, mean, var, g, b, C, relu, M*C);
}

static void run_orl(const float* fl, const int* idx, const float* w1, const float* w2,
                    float* gm, float* fg, float* cc, float* outb,
                    int M, int Npts, int K, int cout, _Float16* wbuf, hipStream_t s)
{
  gathermean_kernel<<<CDIV(M*cout,256), 256, 0, s>>>(fl, idx, gm, Npts, K, cout, M*cout);
  gemm(gm, w1, nullptr, fg, M, cout, cout, 1, 1, wbuf, s);
  concat2_kernel<<<CDIV(M*2*cout,256), 256, 0, s>>>(fl, fg, cc, cout, cout, M*2*cout);
  gemm(cc, w2, nullptr, outb, M, cout, 2*cout, 1, 1, wbuf, s);
}

// ======================= launch =======================
extern "C" void kernel_launch(void* const* d_in, const int* in_sizes, int n_in,
                              void* d_out, int out_size, void* d_ws, size_t ws_size,
                              hipStream_t stream)
{
  (void)in_sizes; (void)n_in; (void)out_size; (void)ws_size;
  const int B = 4, N0 = 4096, N1 = 1024, N2 = 256;
  const int M0 = B*N0, M1 = B*N1, M2 = B*N2;

  const float* verts = (const float*)d_in[0];
  const int*   cat   = (const int*)d_in[1];

  // Param leaves, jax pytree flatten order (dicts sorted by key, lists/tuples in order):
  // blk, bn1, bn2, bn3, c0{dirs,orl1,orl2,ste}, c1..c4{b,dirs,orl1,orl2,ste,w},
  // face, face_out, recon, recon_out
  int pi = 2;
  auto nf = [&]() { return (const float*)d_in[pi++]; };
  const float *blk_w[3], *blk_b[3], *blk_g[3], *blk_be[3];
  for (int i = 0; i < 3; ++i) { blk_w[i]=nf(); blk_b[i]=nf(); blk_g[i]=nf(); blk_be[i]=nf(); }
  const float *bn1_g=nf(), *bn1_b=nf();
  const float *bn2_g=nf(), *bn2_b=nf();
  const float *bn3_g=nf(), *bn3_b=nf();
  const float *c0_dirs=nf(), *c0_o1=nf(), *c0_o2=nf(), *c0_ste=nf();
  const float *c1_b=nf(), *c1_dirs=nf(), *c1_o1=nf(), *c1_o2=nf(), *c1_ste=nf(), *c1_w=nf();
  const float *c2_b=nf(), *c2_dirs=nf(), *c2_o1=nf(), *c2_o2=nf(), *c2_ste=nf(), *c2_w=nf();
  const float *c3_b=nf(), *c3_dirs=nf(), *c3_o1=nf(), *c3_o2=nf(), *c3_ste=nf(), *c3_w=nf();
  const float *c4_b=nf(), *c4_dirs=nf(), *c4_o1=nf(), *c4_o2=nf(), *c4_ste=nf(), *c4_w=nf();
  const float *fc_w[3], *fc_b[3], *fc_g[3], *fc_be[3];
  for (int i = 0; i < 3; ++i) { fc_w[i]=nf(); fc_b[i]=nf(); fc_g[i]=nf(); fc_be[i]=nf(); }
  const float *fo_w=nf(), *fo_b=nf();
  const float *rc_w=nf(), *rc_b=nf(), *rc_g=nf(), *rc_be=nf();
  const float *ro_w=nf(), *ro_b=nf();

  float* outRecon = (float*)d_out;                       // M0 x 3
  float* outFace  = outRecon + (size_t)M0*3;             // M0 x 30
  float* outFeat  = outFace  + (size_t)M0*30;            // M0 x 1286

  // -------- workspace layout: persistent region + reused temp arena --------
  float* Wb = (float*)d_ws;
  size_t off = 0;
  auto alloc = [&](size_t n) -> float* { float* p = Wb + off; off += (n + 255) & ~(size_t)255; return p; };
  float* fm0 = alloc((size_t)M0*128);
  float* fm1 = alloc((size_t)M0*128);
  float* fm2 = alloc((size_t)M1*256);
  float* fm3 = alloc((size_t)M1*256);
  float* fm4 = alloc((size_t)M2*512);
  float* v1  = alloc((size_t)M1*3);
  float* v2  = alloc((size_t)M2*3);
  float* f1  = alloc((size_t)M1*128);
  float* f2  = alloc((size_t)M2*256);
  float* fgl = alloc((size_t)B*512);
  float* mnb = alloc(512);
  float* vrb = alloc(512);
  float* nds = alloc((size_t)3*7*512);
  int* idx0 = (int*)alloc((size_t)M0*10);
  int* idx2 = (int*)alloc((size_t)M1*10);
  int* idx4 = (int*)alloc((size_t)M2*10);
  int* idxp = (int*)alloc((size_t)M0);        // pool knn (<= 4096 rows * 4)
  int* n1i  = (int*)alloc((size_t)M0);
  int* n2i  = (int*)alloc((size_t)M0);
  float* xblk = alloc((size_t)M0*256);        // blk output, alive through face branch
  _Float16* wbuf = (_Float16*)alloc(560000);  // packed-weight scratch (max 4096x256 halves)
  float* T = Wb + off;                        // temp arena (reused per stage)
  hipStream_t s = stream;

  // -------- level-0 kNN (shared by c0 and c1) --------
  knn_kernel<11><<<CDIV(M0,128), 128, 0, s>>>(verts, idx0, B, N0, N0, 1);

  // -------- c0: hs_surface --------
  {
    float* fl = T;                  float* gm = fl + (size_t)M0*128;
    float* fg = gm + (size_t)M0*128; float* cc = fg + (size_t)M0*128;
    ndirs_kernel<<<CDIV(7*128,128), 128, 0, s>>>(c0_dirs, nds, 7*128);
    hs_local_kernel<<<M0, 128, 0, s>>>(verts, idx0, nds, c0_ste, nullptr, fl, N0, 10, 128);
    run_orl(fl, idx0, c0_o1, c0_o2, gm, fg, cc, fm0, M0, N0, 10, 128, wbuf, s);
  }

  // -------- c1: hs_layer + bn1 + relu --------
  {
    float* fo = T;                        float* fl = fo + (size_t)M0*1024;
    float* gm = fl + (size_t)M0*128;      float* fg = gm + (size_t)M0*128;
    float* cc = fg + (size_t)M0*128;      float* raw = cc + (size_t)M0*256;
    gemm(fm0, c1_w, c1_b, fo, M0, 1024, 128, 0, 0, wbuf, s);
    ndirs_kernel<<<CDIV(7*128,128), 128, 0, s>>>(c1_dirs, nds, 7*128);
    hs_local_kernel<<<M0, 128, 0, s>>>(verts, idx0, nds, c1_ste, fo, fl, N0, 10, 128);
    run_orl(fl, idx0, c1_o1, c1_o2, gm, fg, cc, raw, M0, N0, 10, 128, wbuf, s);
    run_bn(raw, fm1, bn1_g, bn1_b, mnb, vrb, M0, 128, 1, s);
  }

  // -------- pool1: vertices -> v1, fm1 -> f1 --------
  knn_kernel<5><<<CDIV(B*N1,128), 128, 0, s>>>(verts, idxp, B, N0, N1, 4);
  poolmax_kernel<<<CDIV(M1*128,256), 256, 0, s>>>(fm1, idxp, f1, N0, 4, 128, N1, M1*128);
  subsample_kernel<<<CDIV(M1*3,256), 256, 0, s>>>(verts, v1, N0, N1, 4, M1*3);

  // -------- level-1 kNN (shared by c2 and c3) --------
  knn_kernel<11><<<CDIV(M1,128), 128, 0, s>>>(v1, idx2, B, N1, N1, 1);

  // -------- c2 --------
  {
    float* fo = T;                        float* fl = fo + (size_t)M1*2048;
    float* gm = fl + (size_t)M1*256;      float* fg = gm + (size_t)M1*256;
    float* cc = fg + (size_t)M1*256;      float* raw = cc + (size_t)M1*512;
    gemm(f1, c2_w, c2_b, fo, M1, 2048, 128, 0, 0, wbuf, s);
    ndirs_kernel<<<CDIV(7*256,128), 128, 0, s>>>(c2_dirs, nds, 7*256);
    hs_local_kernel<<<M1, 128, 0, s>>>(v1, idx2, nds, c2_ste, fo, fl, N1, 10, 256);
    run_orl(fl, idx2, c2_o1, c2_o2, gm, fg, cc, raw, M1, N1, 10, 256, wbuf, s);
    run_bn(raw, fm2, bn2_g, bn2_b, mnb, vrb, M1, 256, 1, s);
  }

  // -------- c3 --------
  {
    float* fo = T;                        float* fl = fo + (size_t)M1*2048;
    float* gm = fl + (size_t)M1*256;      float* fg = gm + (size_t)M1*256;
    float* cc = fg + (size_t)M1*256;      float* raw = cc + (size_t)M1*512;
    gemm(fm2, c3_w, c3_b, fo, M1, 2048, 256, 0, 0, wbuf, s);
    ndirs_kernel<<<CDIV(7*256,128), 128, 0, s>>>(c3_dirs, nds, 7*256);
    hs_local_kernel<<<M1, 128, 0, s>>>(v1, idx2, nds, c3_ste, fo, fl, N1, 10, 256);
    run_orl(fl, idx2, c3_o1, c3_o2, gm, fg, cc, raw, M1, N1, 10, 256, wbuf, s);
    run_bn(raw, fm3, bn3_g, bn3_b, mnb, vrb, M1, 256, 1, s);
  }

  // -------- pool2: v1 -> v2, fm3 -> f2 --------
  knn_kernel<5><<<CDIV(B*N2,128), 128, 0, s>>>(v1, idxp, B, N1, N2, 4);
  poolmax_kernel<<<CDIV(M2*256,256), 256, 0, s>>>(fm3, idxp, f2, N1, 4, 256, N2, M2*256);
  subsample_kernel<<<CDIV(M2*3,256), 256, 0, s>>>(v1, v2, N1, N2, 4, M2*3);

  // -------- c4 (no bn / no extra relu) --------
  knn_kernel<11><<<CDIV(M2,128), 128, 0, s>>>(v2, idx4, B, N2, N2, 1);
  {
    float* fo = T;                        float* fl = fo + (size_t)M2*4096;
    float* gm = fl + (size_t)M2*512;      float* fg = gm + (size_t)M2*512;
    float* cc = fg + (size_t)M2*512;
    gemm(f2, c4_w, c4_b, fo, M2, 4096, 256, 0, 0, wbuf, s);
    ndirs_kernel<<<CDIV(7*512,128), 128, 0, s>>>(c4_dirs, nds, 7*512);
    hs_local_kernel<<<M2, 128, 0, s>>>(v2, idx4, nds, c4_ste, fo, fl, N2, 10, 512);
    run_orl(fl, idx4, c4_o1, c4_o2, gm, fg, cc, fm4, M2, N2, 10, 512, wbuf, s);
  }

  // -------- global max, nearest-neighbor upsample indices, feat concat --------
  gmax_kernel<<<CDIV(B*512,256), 256, 0, s>>>(fm4, fgl, N2, 512, B*512);
  nearest_kernel<<<CDIV(M0,256), 256, 0, s>>>(verts, v1, n1i, N0, N1, M0);
  nearest_kernel<<<CDIV(M0,256), 256, 0, s>>>(verts, v2, n2i, N0, N2, M0);
  feat_kernel<<<CDIV(M0*1286,256), 256, 0, s>>>(fm0, fm1, fm2, fm3, fm4, n1i, n2i, cat,
                                                outFeat, N0, N1, N2, M0*1286);

  // -------- blk MLP: 1286 -> 512 -> 512 -> 256 (conv+bias, bn, relu) --------
  {
    float* b0 = T; float* b1 = b0 + (size_t)M0*512;
    gemm(outFeat, blk_w[0], blk_b[0], b0, M0, 512, 1286, 1, 0, wbuf, s);
    run_bn(b0, b0, blk_g[0], blk_be[0], mnb, vrb, M0, 512, 1, s);
    gemm(b0, blk_w[1], blk_b[1], b1, M0, 512, 512, 1, 0, wbuf, s);
    run_bn(b1, b1, blk_g[1], blk_be[1], mnb, vrb, M0, 512, 1, s);
    gemm(b1, blk_w[2], blk_b[2], xblk, M0, 256, 512, 1, 0, wbuf, s);
    run_bn(xblk, xblk, blk_g[2], blk_be[2], mnb, vrb, M0, 256, 1, s);
  }

  // -------- recon branch --------
  {
    float* r = T;
    gemm(xblk, rc_w, rc_b, r, M0, 128, 256, 1, 0, wbuf, s);
    run_bn(r, r, rc_g, rc_be, mnb, vrb, M0, 128, 1, s);
    gemm(r, ro_w, ro_b, outRecon, M0, 3, 128, 1, 0, wbuf, s);
  }

  // -------- face branch --------
  {
    float* fin = T;                      float* y0 = fin + (size_t)M0*771;
    float* y1  = y0 + (size_t)M0*512;    float* y2 = y1 + (size_t)M0*256;
    facein_kernel<<<CDIV(M0*771,256), 256, 0, s>>>(fgl, xblk, verts, fin, N0, M0*771);
    gemm(fin, fc_w[0], fc_b[0], y0, M0, 512, 771, 1, 0, wbuf, s);
    run_bn(y0, y0, fc_g[0], fc_be[0], mnb, vrb, M0, 512, 1, s);
    gemm(y0, fc_w[1], fc_b[1], y1, M0, 256, 512, 1, 0, wbuf, s);
    run_bn(y1, y1, fc_g[1], fc_be[1], mnb, vrb, M0, 256, 1, s);
    gemm(y1, fc_w[2], fc_b[2], y2, M0, 128, 256, 1, 0, wbuf, s);
    run_bn(y2, y2, fc_g[2], fc_be[2], mnb, vrb, M0, 128, 1, s);
    gemm(y2, fo_w, fo_b, outFace, M0, 30, 128, 1, 0, wbuf, s);
  }
}
